// MaskedContextSelector_8306466750964
// MI455X (gfx1250) — compile-verified, matrix operand order checked by reading, not run on hardware
//
#include <hip/hip_runtime.h>
#include <math.h>

#define DEVINL __device__ __forceinline__

typedef __attribute__((ext_vector_type(16))) __bf16 v16bf;
typedef __attribute__((ext_vector_type(8)))  float  v8f;
typedef __attribute__((ext_vector_type(4)))  unsigned int u32x4;
typedef __attribute__((ext_vector_type(8)))  int i32x8;
typedef __attribute__((ext_vector_type(4)))  int i32x4;

// pointer types for the async global->LDS builtin (param types per hipcc
// diagnostic: v4i __device__* src, v4i __shared__* dst)
typedef int v4i_t __attribute__((ext_vector_type(4)));
typedef v4i_t __attribute__((address_space(1)))* as1_v4i;
typedef v4i_t __attribute__((address_space(3)))* as3_v4i;

#ifndef __has_builtin
#define __has_builtin(x) 0
#endif
#if __has_builtin(__builtin_amdgcn_global_load_async_to_lds_b128) && \
    __has_builtin(__builtin_amdgcn_s_wait_asynccnt)
#define USE_ASYNC 1
#else
#define USE_ASYNC 0
#endif
#if __has_builtin(__builtin_amdgcn_tensor_load_to_lds) && \
    __has_builtin(__builtin_amdgcn_s_wait_tensorcnt)
#define USE_TDM 1
#else
#define USE_TDM 0
#endif

constexpr int B  = 2;
constexpr int T  = 2048;
constexpr int E  = 1024;
constexpr int NH = 16;
constexpr int DH = 64;
constexpr int S  = 2048;
constexpr float SCALE = 0.125f;   // 1/sqrt(64)
constexpr float INV_TAU = 0.5f;   // 1/2.0
constexpr float NEGV = -1e9f;
constexpr int SP = S + 8;         // padded attn row

// block-tiled GEMM geometry
constexpr int BM = 128, BN = 128, BK = 64;
constexpr int LDT = BK + 8;       // 72-elem padded LDS row (bank spread, 16B rows)

// ---- CDNA5 16-bit fragment helpers -----------------------------------------
DEVINL int kpair(int v, int half) {
  return (v < 4 ? 2 * v : 16 + 2 * (v - 4)) + 8 * half;
}

// 16x32 bf16 fragment, K contiguous in memory with leading dim `ld` (elems).
DEVINL v16bf load_frag_k32(const __bf16* __restrict__ src, int row, int ld,
                           int kb, int half) {
  v16bf f;
  const __bf16* p = src + (size_t)row * ld + kb;
#pragma unroll
  for (int v = 0; v < 8; ++v) {
    int k = kpair(v, half);
    f[2 * v]     = p[k];
    f[2 * v + 1] = p[k + 1];
  }
  return f;
}

DEVINL v8f wmma_bf16(v16bf a, v16bf b, v8f c) {
  return __builtin_amdgcn_wmma_f32_16x16x32_bf16(false, a, false, b,
                                                 (short)0, c, false, false);
}

// ---- async panel staging: 128 rows x 64 bf16 (128B/row), 256 threads -------
DEVINL void stage_panel(const __bf16* __restrict__ gsrc, int row0, int lda,
                        int kb, __bf16* lds, int tid) {
#pragma unroll
  for (int i = 0; i < 4; ++i) {
    int chunk = tid + i * 256;     // 1024 chunks of 16B
    int row = chunk >> 3;
    int jb  = (chunk & 7) * 8;
    const __bf16* g = gsrc + (size_t)(row0 + row) * lda + kb + jb;
    __bf16* d = lds + row * LDT + jb;
#if USE_ASYNC
    __builtin_amdgcn_global_load_async_to_lds_b128(
        (as1_v4i)(uintptr_t)g, (as3_v4i)(unsigned)(uintptr_t)d, 0, 0);
#else
    *(float4*)d = *(const float4*)g;
#endif
  }
}

// ---- shared block-GEMM core: C[128x128] = A[128xK] * B[128xK]^T ------------
// acc[4][2]: each wave (of 8) owns a 64x32 region (4 M-tiles x 2 N-tiles).
DEVINL void gemm_block(const __bf16* __restrict__ Ag, int lda,
                       const __bf16* __restrict__ Bg, int ldb,
                       int mBlk, int nBlk, int K,
                       __bf16* AsmP, __bf16* BsmP, v8f acc[4][2]) {
  int tid = threadIdx.x;
  int wave = tid >> 5, lane = tid & 31;
  int half = lane >> 4, col = lane & 15;
  int wm = (wave & 1) * 64, wn = (wave >> 1) * 32;
  const int PSZ = BM * LDT;

  stage_panel(Ag, mBlk, lda, 0, AsmP, tid);
  stage_panel(Bg, nBlk, ldb, 0, BsmP, tid);
#if USE_ASYNC
  __builtin_amdgcn_s_wait_asynccnt(0);
#endif
  __syncthreads();

  for (int kb = 0; kb < K; kb += BK) {
    int buf = (kb / BK) & 1;
    if (kb + BK < K) {  // overlap: issue next panel while computing this one
      stage_panel(Ag, mBlk, lda, kb + BK, AsmP + (buf ^ 1) * PSZ, tid);
      stage_panel(Bg, nBlk, ldb, kb + BK, BsmP + (buf ^ 1) * PSZ, tid);
    }
    const __bf16* As = AsmP + buf * PSZ;
    const __bf16* Bs = BsmP + buf * PSZ;
#pragma unroll
    for (int ks = 0; ks < BK; ks += 32) {
      v16bf af[4], bf[2];
#pragma unroll
      for (int mt = 0; mt < 4; ++mt)
        af[mt] = load_frag_k32(As, wm + mt * 16 + col, LDT, ks, half);
#pragma unroll
      for (int nt = 0; nt < 2; ++nt)
        bf[nt] = load_frag_k32(Bs, wn + nt * 16 + col, LDT, ks, half);
#pragma unroll
      for (int mt = 0; mt < 4; ++mt)
#pragma unroll
        for (int nt = 0; nt < 2; ++nt)
          acc[mt][nt] = wmma_bf16(af[mt], bf[nt], acc[mt][nt]);
    }
#if USE_ASYNC
    __builtin_amdgcn_s_wait_asynccnt(0);
#endif
    __syncthreads();
  }
}

// ---- cast kernels ----------------------------------------------------------
__global__ void cast_hs_kernel(const float* __restrict__ hs,
                               __bf16* __restrict__ hs_bf,
                               __bf16* __restrict__ hsT_bf) {
  size_t i = (size_t)blockIdx.x * 256 + threadIdx.x;
  if (i >= (size_t)B * T * E) return;
  __bf16 bv = (__bf16)hs[i];
  hs_bf[i] = bv;
  int e = (int)(i % E);
  size_t bt = i / E;
  int t = (int)(bt % T);
  int bb = (int)(bt / T);
  hsT_bf[((size_t)bb * E + e) * T + t] = bv;
}

__global__ void cast_mat_kernel(const float* __restrict__ w,
                                __bf16* __restrict__ wb, int n) {
  int i = blockIdx.x * 256 + threadIdx.x;
  if (i < n) wb[i] = (__bf16)w[i];
}

// ---- Q/K projection, block-tiled -------------------------------------------
__global__ void __launch_bounds__(256) qk_proj_kernel(
    const __bf16* __restrict__ hs_bf,
    const __bf16* __restrict__ Wq, const float* __restrict__ bq,
    const __bf16* __restrict__ Wk, const float* __restrict__ bk,
    __bf16* __restrict__ Qo, __bf16* __restrict__ Ko) {
  __shared__ __bf16 Asm[2][BM][LDT];
  __shared__ __bf16 Bsm[2][BM][LDT];
  const int MB = (B * T) / BM;   // 32
  const int NB = E / BN;         // 8
  int blk = blockIdx.x;
  int mat = blk / (MB * NB);
  int rem = blk % (MB * NB);
  int mBlk = (rem / NB) * BM;
  int nBlk = (rem % NB) * BN;
  const __bf16* Wmat = mat ? Wk : Wq;
  const float*  bias = mat ? bk : bq;
  __bf16*       Out  = mat ? Ko : Qo;

  v8f acc[4][2] = {};
  gemm_block(hs_bf, E, Wmat, E, mBlk, nBlk, E, &Asm[0][0][0], &Bsm[0][0][0],
             acc);

  int lane = threadIdx.x & 31, wave = threadIdx.x >> 5;
  int half = lane >> 4, col = lane & 15;
  int wm = (wave & 1) * 64, wn = (wave >> 1) * 32;
#pragma unroll
  for (int mt = 0; mt < 4; ++mt)
#pragma unroll
    for (int nt = 0; nt < 2; ++nt) {
      int n = nBlk + wn + nt * 16 + col;
      float bv = bias[n];
#pragma unroll
      for (int rr = 0; rr < 8; ++rr) {
        int m = mBlk + wm + mt * 16 + rr + 8 * half;
        Out[(size_t)m * E + n] = (__bf16)(acc[mt][nt][rr] + bv);
      }
    }
}

// ---- fused attention -------------------------------------------------------
__global__ void __launch_bounds__(256) attn_kernel(
    const __bf16* __restrict__ Qg, const __bf16* __restrict__ Kg,
    const float* __restrict__ gumbel_u, const int* __restrict__ lens,
    __bf16* __restrict__ dwo) {
  __shared__ __bf16 q_lds[NH][16][DH];   // 32 KB, head-major
  __shared__ int    len_lds[16];
  __shared__ float2 ml_lds[NH][16];
  __shared__ float  attn_lds[16][SP];    // ~128.5 KB

  int bb = blockIdx.x / (T / 16);
  int t0 = (blockIdx.x % (T / 16)) * 16;
  int tid = threadIdx.x;
  int wave = tid >> 5, lane = tid & 31;
  int half = lane >> 4, col = lane & 15;

  if (tid < 16) len_lds[tid] = lens[t0 + tid];

#if USE_TDM
  // TDM iterate-mode staging: 16 iterations of a 64x16 tile with
  // global_addr_increment = 64 elems (next head) and lds_addr_increment =
  // 1024 elems performs the [t][h][d] -> [h][t][d] transpose in one DMA.
  if (wave == 0) {
    unsigned lds_addr = (unsigned)(uintptr_t)(&q_lds[0][0][0]);
    unsigned long long ga =
        (unsigned long long)(uintptr_t)(Qg + ((size_t)bb * T + t0) * NH * DH);
    u32x4 g0 = {1u,                                   // count = 1
                lds_addr,
                (unsigned)(ga & 0xFFFFFFFFu),
                (unsigned)((ga >> 32) & 0x01FFFFFFu) | (2u << 30)};  // type=2
    i32x8 g1 = {(1 << 16) | (1 << 19),   // data_size=2B, iterate_enable
                (int)(1024u << 16),      // tensor_dim0 = 1024 (lo16 in [31:16])
                (int)(16u << 16),        // tensor_dim1 = 16
                (int)(64u << 16),        // tile_dim0 = 64
                16,                      // tile_dim1 = 16, tile_dim2 = 0
                1024,                    // tensor_dim0_stride (elems)
                0, 0};
    i32x4 g2 = {0,                       // tensor_dim2
                1024,                    // lds_addr_increment (elems)
                64,                      // global_addr_increment (elems)
                (15 << 16)};             // iterate_count = 15 -> 16 iterations
    i32x4 g3 = {0, 0, 0, 0};
#if __clang_major__ >= 23
    i32x8 g4 = {0, 0, 0, 0, 0, 0, 0, 0};
    __builtin_amdgcn_tensor_load_to_lds(g0, g1, g2, g3, g4, 0);
#else
    __builtin_amdgcn_tensor_load_to_lds(g0, g1, g2, g3, 0);
#endif
    __builtin_amdgcn_s_wait_tensorcnt(0);
  }
#else
  {  // fallback: thread tid -> (t, h) row of 64 bf16 (8x float4)
    int tq = tid >> 4, h = tid & 15;
    const float4* s4 =
        (const float4*)(Qg + (((size_t)bb * T + t0 + tq) * NH + h) * DH);
    float4* d4 = (float4*)&q_lds[h][tq][0];
#pragma unroll
    for (int i = 0; i < 8; ++i) d4[i] = s4[i];
  }
#endif
  __syncthreads();

  const __bf16* Kbase = Kg + (size_t)bb * T * NH * DH;

  // phase 1: online (rowmax, expsum) per (head,row); 2 heads per wave
  for (int hh = 0; hh < 2; ++hh) {
    int h = wave * 2 + hh;
    v16bf a0 = load_frag_k32(&q_lds[h][0][0], col, DH, 0,  half);
    v16bf a1 = load_frag_k32(&q_lds[h][0][0], col, DH, 32, half);
    float m_r[8], l_r[8];
#pragma unroll
    for (int rr = 0; rr < 8; ++rr) { m_r[rr] = -3.0e38f; l_r[rr] = 0.f; }
    for (int sb = 0; sb < S / 16; ++sb) {
      int s0 = sb * 16;
      if (sb + 1 < S / 16)  // pull next K panel toward L2/L0
        __builtin_prefetch(Kbase + h * DH + (size_t)(s0 + 16 + col) * NH * DH,
                           0, 1);
      v16bf b0 = load_frag_k32(Kbase + h * DH, s0 + col, NH * DH, 0,  half);
      v16bf b1 = load_frag_k32(Kbase + h * DH, s0 + col, NH * DH, 32, half);
      v8f accv = {};
      accv = wmma_bf16(a0, b0, accv);
      accv = wmma_bf16(a1, b1, accv);
      int s = s0 + col;
#pragma unroll
      for (int rr = 0; rr < 8; ++rr) {
        int trow = rr + 8 * half;
        float x = (s < len_lds[trow]) ? accv[rr] * SCALE : NEGV;
        float tm = x;
        tm = fmaxf(tm, __shfl_xor(tm, 1, 32));
        tm = fmaxf(tm, __shfl_xor(tm, 2, 32));
        tm = fmaxf(tm, __shfl_xor(tm, 4, 32));
        tm = fmaxf(tm, __shfl_xor(tm, 8, 32));
        float mnew = fmaxf(m_r[rr], tm);
        float e = __expf(x - mnew);
        e += __shfl_xor(e, 1, 32);
        e += __shfl_xor(e, 2, 32);
        e += __shfl_xor(e, 4, 32);
        e += __shfl_xor(e, 8, 32);
        l_r[rr] = l_r[rr] * __expf(m_r[rr] - mnew) + e;
        m_r[rr] = mnew;
      }
    }
    if (col == 0) {
#pragma unroll
      for (int rr = 0; rr < 8; ++rr)
        ml_lds[h][rr + 8 * half] = make_float2(m_r[rr], l_r[rr]);
    }
  }
  __syncthreads();

  // phase 2: head-averaged probabilities; wave owns s in [wave*256, +256)
  const float inv_nh = 1.0f / NH;
  for (int sbl = 0; sbl < 16; ++sbl) {
    int s0 = wave * 256 + sbl * 16;
    float pacc[8];
#pragma unroll
    for (int rr = 0; rr < 8; ++rr) pacc[rr] = 0.f;
    for (int h = 0; h < NH; ++h) {
      v16bf a0 = load_frag_k32(&q_lds[h][0][0], col, DH, 0,  half);
      v16bf a1 = load_frag_k32(&q_lds[h][0][0], col, DH, 32, half);
      v16bf b0 = load_frag_k32(Kbase + h * DH, s0 + col, NH * DH, 0,  half);
      v16bf b1 = load_frag_k32(Kbase + h * DH, s0 + col, NH * DH, 32, half);
      v8f accv = {};
      accv = wmma_bf16(a0, b0, accv);
      accv = wmma_bf16(a1, b1, accv);
      int s = s0 + col;
#pragma unroll
      for (int rr = 0; rr < 8; ++rr) {
        int trow = rr + 8 * half;
        float2 ml = ml_lds[h][trow];
        float p = (s < len_lds[trow])
                      ? __expf(accv[rr] * SCALE - ml.x) / ml.y : 0.f;
        pacc[rr] += p;
      }
    }
#pragma unroll
    for (int rr = 0; rr < 8; ++rr)
      attn_lds[rr + 8 * half][s0 + col] = pacc[rr] * inv_nh;
  }
  __syncthreads();

  // phase 3: gumbel + ragged-masked softmax; wave owns rows 2w, 2w+1
  for (int tl = wave * 2; tl < wave * 2 + 2; ++tl) {
    int t = t0 + tl;
    int len = len_lds[tl];
    const float* urow = gumbel_u + ((size_t)bb * T + t) * S;
    float mx = -3.0e38f;
    for (int s = lane; s < S; s += 32) {
      float lg;
      if (s < len) {
        float g = -__logf(-__logf(urow[s]));
        lg = (attn_lds[tl][s] + g) * INV_TAU;
      } else {
        lg = NEGV;
      }
      attn_lds[tl][s] = lg;
      mx = fmaxf(mx, lg);
    }
    mx = fmaxf(mx, __shfl_xor(mx, 1, 32));
    mx = fmaxf(mx, __shfl_xor(mx, 2, 32));
    mx = fmaxf(mx, __shfl_xor(mx, 4, 32));
    mx = fmaxf(mx, __shfl_xor(mx, 8, 32));
    mx = fmaxf(mx, __shfl_xor(mx, 16, 32));
    float sum = 0.f;
    for (int s = lane; s < S; s += 32) sum += __expf(attn_lds[tl][s] - mx);
    sum += __shfl_xor(sum, 1, 32);
    sum += __shfl_xor(sum, 2, 32);
    sum += __shfl_xor(sum, 4, 32);
    sum += __shfl_xor(sum, 8, 32);
    sum += __shfl_xor(sum, 16, 32);
    float inv = 1.0f / sum;
    __bf16* drow = dwo + ((size_t)bb * T + t) * S;
    for (int s = lane; s < S; s += 32)
      drow[s] = (__bf16)(__expf(attn_lds[tl][s] - mx) * inv);
  }
}

// ---- output GEMM, block-tiled ----------------------------------------------
__global__ void __launch_bounds__(256) out_gemm_kernel(
    const __bf16* __restrict__ dw, const __bf16* __restrict__ hsT,
    const int* __restrict__ lens, float* __restrict__ out) {
  __shared__ __bf16 Asm[2][BM][LDT];
  __shared__ __bf16 Bsm[2][BM][LDT];
  const int MB = T / BM;   // 16
  const int NB = E / BN;   // 8
  int blk = blockIdx.x;
  int bb  = blk / (MB * NB);
  int rem = blk % (MB * NB);
  int mBlk = (rem / NB) * BM;
  int nBlk = (rem % NB) * BN;
  const __bf16* A  = dw  + (size_t)bb * T * S;   // [t][s]
  const __bf16* Bm = hsT + (size_t)bb * E * T;   // [e][s]

  v8f acc[4][2] = {};
  gemm_block(A, S, Bm, T, mBlk, nBlk, S, &Asm[0][0][0], &Bsm[0][0][0], acc);

  int lane = threadIdx.x & 31, wave = threadIdx.x >> 5;
  int half = lane >> 4, col = lane & 15;
  int wm = (wave & 1) * 64, wn = (wave >> 1) * 32;
#pragma unroll
  for (int mt = 0; mt < 4; ++mt)
#pragma unroll
    for (int rr = 0; rr < 8; ++rr) {
      int t = mBlk + wm + mt * 16 + rr + 8 * half;
      float valid = (lens[t] > 0) ? 1.f : 0.f;
#pragma unroll
      for (int nt = 0; nt < 2; ++nt) {
        int n = nBlk + wn + nt * 16 + col;
        out[((size_t)bb * T + t) * E + n] = acc[mt][nt][rr] * valid;
      }
    }
}

// ---- host ------------------------------------------------------------------
extern "C" void kernel_launch(void* const* d_in, const int* in_sizes, int n_in,
                              void* d_out, int out_size, void* d_ws,
                              size_t ws_size, hipStream_t stream) {
  (void)in_sizes; (void)n_in; (void)out_size;
  const float* hs   = (const float*)d_in[0];
  const float* Wq   = (const float*)d_in[1];
  const float* bq   = (const float*)d_in[2];
  const float* Wk   = (const float*)d_in[3];
  const float* bk   = (const float*)d_in[4];
  const float* gu   = (const float*)d_in[5];
  const int*   lens = (const int*)d_in[6];

  char* ws = (char*)d_ws;
  size_t off = 0;
  auto alloc = [&](size_t bytes) -> char* {
    char* p = ws + off;
    off += (bytes + 255) & ~(size_t)255;
    return p;
  };
  __bf16* hs_bf  = (__bf16*)alloc((size_t)B * T * E * 2);  // 8 MB
  __bf16* hsT_bf = (__bf16*)alloc((size_t)B * T * E * 2);  // 8 MB
  __bf16* Wq_bf  = (__bf16*)alloc((size_t)E * E * 2);      // 2 MB
  __bf16* Wk_bf  = (__bf16*)alloc((size_t)E * E * 2);      // 2 MB
  __bf16* Qb     = (__bf16*)alloc((size_t)B * T * E * 2);  // 8 MB
  __bf16* Kb     = (__bf16*)alloc((size_t)B * T * E * 2);  // 8 MB
  __bf16* dwb    = (__bf16*)alloc((size_t)B * T * S * 2);  // 16 MB
  if (off > ws_size) return;

  cast_hs_kernel<<<(B * T * E + 255) / 256, 256, 0, stream>>>(hs, hs_bf,
                                                              hsT_bf);
  cast_mat_kernel<<<(E * E + 255) / 256, 256, 0, stream>>>(Wq, Wq_bf, E * E);
  cast_mat_kernel<<<(E * E + 255) / 256, 256, 0, stream>>>(Wk, Wk_bf, E * E);

  // 2 matrices x (4096/128) x (1024/128) = 512 blocks
  qk_proj_kernel<<<2 * ((B * T) / BM) * (E / BN), 256, 0, stream>>>(
      hs_bf, Wq_bf, bq, Wk_bf, bk, Qb, Kb);
  attn_kernel<<<B * (T / 16), 256, 0, stream>>>(Qb, Kb, gu, lens, dwb);
  // 2 x (2048/128) x (1024/128) = 256 blocks
  out_gemm_kernel<<<B * (T / BM) * (E / BN), 256, 0, stream>>>(
      dwb, hsT_bf, lens, (float*)d_out);
}
